// DLinear_2113123910009
// MI455X (gfx1250) — matrix operand order repaired and checked by reading, not compiled, not run on hardware
//
#include <hip/hip_runtime.h>

typedef __attribute__((ext_vector_type(16))) __bf16 v16bf;
typedef __attribute__((ext_vector_type(8)))  __bf16 v8bf;
typedef __attribute__((ext_vector_type(8)))  float  v8f;
typedef __attribute__((ext_vector_type(4)))  float  v4f;
typedef __attribute__((ext_vector_type(4)))  int    v4i;

#define SEQ    512
#define PRED   96
#define CIN    128
#define NBATCH 512
#define BM     64
#define KT     32
#define ASTR   40   // halfs per A-tile row in LDS (80B: conflict-friendly, 16B aligned)
#define BSTR   40   // halfs per B-tile row in LDS

// ---- CDNA5 async memory->LDS path (guarded; falls back to sync copy) ------
#if defined(__has_builtin)
#  if __has_builtin(__builtin_amdgcn_global_load_async_to_lds_b128)
#    define ASYNC_B 1
#  endif
#endif
#ifndef ASYNC_B
#  define ASYNC_B 0
#endif

__device__ __forceinline__ void wait_async0() {
#if defined(__has_builtin) && __has_builtin(__builtin_amdgcn_s_wait_asynccnt)
  __builtin_amdgcn_s_wait_asynccnt(0);
#else
  asm volatile("s_wait_asynccnt 0" ::: "memory");
#endif
}

__device__ __forceinline__ void copy16_to_lds(const __bf16* gsrc, __bf16* ldst) {
#if ASYNC_B
  __builtin_amdgcn_global_load_async_to_lds_b128(
      (__attribute__((address_space(1))) v4i*)gsrc,
      (__attribute__((address_space(3))) v4i*)ldst, 0, 0);
#else
  *(v8bf*)ldst = *(const v8bf*)gsrc;
#endif
}

// Stage one 96x32-half B tile (4x16B chunks per p-row) into LDS.
__device__ __forceinline__ void stageB(const __bf16* __restrict__ src,
                                       __bf16* dst, int c, int k0, int tid) {
  for (int idx = tid; idx < PRED * 4; idx += 256) {   // wave-uniform trip count
    const int p = idx >> 2, ch = idx & 3;
    copy16_to_lds(&src[((size_t)c * PRED + p) * SEQ + k0 + ch * 8],
                  &dst[p * BSTR + ch * 8]);
  }
}

union U16 { v16bf v; v8bf h[2]; };

__device__ __forceinline__ v16bf ld16(const __bf16* p) {
  U16 u;
  u.h[0] = *(const v8bf*)(p);
  u.h[1] = *(const v8bf*)(p + 8);
  return u.v;
}

__device__ __forceinline__ v8f wmma_bf16(v16bf a, v16bf b, v8f c) {
  return __builtin_amdgcn_wmma_f32_16x16x32_bf16(false, a, false, b, (short)0, c, false, false);
}

// ---------------------------------------------------------------------------
// K1: weight prep.  Ws,Wt are [C][S][P] fp32.  Produce bf16 hi/lo of Ws and
// Wd = Wt - Ws, transposed to [C][P][S] so the GEMM's B-tile loads are
// contiguous along K.
// ---------------------------------------------------------------------------
__global__ __launch_bounds__(256) void k1_prep_weights(
    const float* __restrict__ Ws, const float* __restrict__ Wt,
    __bf16* __restrict__ WsT_hi, __bf16* __restrict__ WsT_lo,
    __bf16* __restrict__ WdT_hi, __bf16* __restrict__ WdT_lo) {
  __shared__ float ts[64 * 97];
  __shared__ float td[64 * 97];
  const int c  = blockIdx.y;
  const int s0 = blockIdx.x * 64;
  const int tid = threadIdx.x;

  for (int e = tid; e < 64 * PRED; e += 256) {
    const int i = e / PRED, p = e - i * PRED;
    const size_t g = ((size_t)c * SEQ + s0 + i) * PRED + p;
    const float a = Ws[g];
    ts[i * 97 + p] = a;
    td[i * 97 + p] = Wt[g] - a;
  }
  __syncthreads();
  for (int o = tid; o < 64 * PRED; o += 256) {
    const int p = o >> 6, u = o & 63;
    const float a = ts[u * 97 + p];
    const float d = td[u * 97 + p];
    const size_t g = ((size_t)c * PRED + p) * SEQ + s0 + u;
    __bf16 h;
    h = (__bf16)a; WsT_hi[g] = h; WsT_lo[g] = (__bf16)(a - (float)h);
    h = (__bf16)d; WdT_hi[g] = h; WdT_lo[g] = (__bf16)(d - (float)h);
  }
}

// ---------------------------------------------------------------------------
// K2: transpose x [B][S][C] fp32 -> xT [C][B][S] fp32 (coalesced both sides).
// ---------------------------------------------------------------------------
__global__ __launch_bounds__(256) void k2_transpose_x(
    const float* __restrict__ x, float* __restrict__ xT) {
  __shared__ float t[64 * 129];
  const int b  = blockIdx.y;
  const int s0 = blockIdx.x * 64;
  const int tid = threadIdx.x;

  for (int e = tid; e < 64 * CIN; e += 256) {
    const int i = e >> 7, cc = e & 127;
    t[i * 129 + cc] = x[((size_t)b * SEQ + s0 + i) * CIN + cc];
  }
  __syncthreads();
  for (int o = tid; o < 64 * CIN; o += 256) {
    const int cc = o >> 6, u = o & 63;
    xT[((size_t)cc * NBATCH + b) * SEQ + s0 + u] = t[u * 129 + cc];
  }
}

// ---------------------------------------------------------------------------
// K3: fused trend + dual GEMM via bf16x3 WMMA.
// One WG = (channel c, 64-row block).  out = x@Ws + trend@Wd + biases.
// B tiles are staged with async global->LDS copies (ASYNCcnt) issued before
// the VALU-heavy A-stage so the DMA overlaps the moving-average computation.
// ---------------------------------------------------------------------------
__global__ __launch_bounds__(256) void k3_dlinear_gemm(
    const float* __restrict__ xT,
    const __bf16* __restrict__ Bsh, const __bf16* __restrict__ Bsl,
    const __bf16* __restrict__ Bdh, const __bf16* __restrict__ Bdl,
    const float* __restrict__ bias_s, const float* __restrict__ bias_t,
    float* __restrict__ out) {
  __shared__ __bf16 sAxh[BM * ASTR], sAxl[BM * ASTR];
  __shared__ __bf16 sAth[BM * ASTR], sAtl[BM * ASTR];
  __shared__ __bf16 sBsh[PRED * BSTR], sBsl[PRED * BSTR];
  __shared__ __bf16 sBdh[PRED * BSTR], sBdl[PRED * BSTR];

  const int c    = blockIdx.y;
  const int B0   = blockIdx.x * BM;
  const int tid  = threadIdx.x;
  const int lane = tid & 31;
  const int w    = tid >> 5;
  const int mt   = w & 3;          // wave's M-tile within the 64-row block
  const int nb   = (w >> 2) * 3;   // wave's first N-tile (0 or 3)
  const int half = lane >> 4;
  const int l16  = lane & 15;

  // A-staging mapping: thread -> (row, k-segment of 8)
  const int tr   = tid >> 2;
  const int seg  = tid & 3;
  // contiguous k order {0..15 | 16..31}, matching the B tiles:
  const int slot = seg;

  const float* xrow = xT + ((size_t)c * NBATCH + (B0 + tr)) * SEQ;

  v8f acc0 = {}, acc1 = {}, acc2 = {};

#pragma unroll 1
  for (int k0 = 0; k0 < SEQ; k0 += KT) {
    // -------- stage B: async global->LDS (overlaps the A-stage below) ------
    stageB(Bsh, sBsh, c, k0, tid);
    stageB(Bsl, sBsl, c, k0, tid);
    stageB(Bdh, sBdh, c, k0, tid);
    stageB(Bdl, sBdl, c, k0, tid);

    // -------- prefetch next iteration's A window (2 cachelines) --------
    if (k0 + KT < SEQ) {
      const float* pf = xrow + (k0 + KT + seg * 8 - 16);
      __builtin_prefetch(pf, 0, 1);
      __builtin_prefetch(pf + 32, 0, 1);
    }

    // -------- stage A: load window, 25-tap moving average, bf16 split ------
    float wv[32];
    if (k0 >= 32 && k0 <= SEQ - 64) {
      // interior: aligned vector loads of the 40-float superset window
      float wb[40];
      const float* pal = xrow + (k0 + seg * 8 - 16);  // 32B aligned
#pragma unroll
      for (int q = 0; q < 10; ++q) *(v4f*)&wb[q * 4] = *(const v4f*)(pal + q * 4);
#pragma unroll
      for (int u = 0; u < 32; ++u) wv[u] = wb[u + 4];
    } else {
      const int jb = k0 + seg * 8 - 12;
#pragma unroll
      for (int u = 0; u < 32; ++u) {
        int j = jb + u;
        j = j < 0 ? 0 : (j > SEQ - 1 ? SEQ - 1 : j);
        wv[u] = xrow[j];
      }
    }
    float run = 0.f;
#pragma unroll
    for (int u = 0; u < 25; ++u) run += wv[u];
    v8bf xh, xl, th, tl;
#pragma unroll
    for (int i = 0; i < 8; ++i) {
      const float xv = wv[12 + i];
      const float tv = run * (1.0f / 25.0f);
      run += wv[25 + i] - wv[i];
      __bf16 hh = (__bf16)xv;
      xh[i] = hh; xl[i] = (__bf16)(xv - (float)hh);
      __bf16 gg = (__bf16)tv;
      th[i] = gg; tl[i] = (__bf16)(tv - (float)gg);
    }
    const int ao = tr * ASTR + slot * 8;
    *(v8bf*)&sAxh[ao] = xh;
    *(v8bf*)&sAxl[ao] = xl;
    *(v8bf*)&sAth[ao] = th;
    *(v8bf*)&sAtl[ao] = tl;

#if ASYNC_B
    wait_async0();   // all of this wave's async B chunks have landed in LDS
#endif
    __syncthreads();

    // -------- compute: 18 WMMAs / wave / K-step --------
    const int ar = mt * 16 + l16;
    const int ka = half * 16;
    const v16bf Axh = ld16(&sAxh[ar * ASTR + ka]);
    const v16bf Axl = ld16(&sAxl[ar * ASTR + ka]);
    const v16bf Ath = ld16(&sAth[ar * ASTR + ka]);
    const v16bf Atl = ld16(&sAtl[ar * ASTR + ka]);

    v16bf Wsh[3], Wsl[3], Wdh[3], Wdl[3];
#pragma unroll
    for (int n = 0; n < 3; ++n) {
      const int bo = ((nb + n) * 16 + l16) * BSTR + ka;
      Wsh[n] = ld16(&sBsh[bo]);
      Wsl[n] = ld16(&sBsl[bo]);
      Wdh[n] = ld16(&sBdh[bo]);
      Wdl[n] = ld16(&sBdl[bo]);
    }

    acc0 = wmma_bf16(Axh, Wsh[0], acc0);
    acc1 = wmma_bf16(Axh, Wsh[1], acc1);
    acc2 = wmma_bf16(Axh, Wsh[2], acc2);
    acc0 = wmma_bf16(Axl, Wsh[0], acc0);
    acc1 = wmma_bf16(Axl, Wsh[1], acc1);
    acc2 = wmma_bf16(Axl, Wsh[2], acc2);
    acc0 = wmma_bf16(Axh, Wsl[0], acc0);
    acc1 = wmma_bf16(Axh, Wsl[1], acc1);
    acc2 = wmma_bf16(Axh, Wsl[2], acc2);
    acc0 = wmma_bf16(Ath, Wdh[0], acc0);
    acc1 = wmma_bf16(Ath, Wdh[1], acc1);
    acc2 = wmma_bf16(Ath, Wdh[2], acc2);
    acc0 = wmma_bf16(Atl, Wdh[0], acc0);
    acc1 = wmma_bf16(Atl, Wdh[1], acc1);
    acc2 = wmma_bf16(Atl, Wdh[2], acc2);
    acc0 = wmma_bf16(Ath, Wdl[0], acc0);
    acc1 = wmma_bf16(Ath, Wdl[1], acc1);
    acc2 = wmma_bf16(Ath, Wdl[2], acc2);
    __syncthreads();
  }

  // -------- epilogue: biases + scatter to [B][P][C] --------
  v8f accs[3] = {acc0, acc1, acc2};
#pragma unroll
  for (int n = 0; n < 3; ++n) {
    const int p = (nb + n) * 16 + l16;
    const float bias = bias_s[c * PRED + p] + bias_t[c * PRED + p];
#pragma unroll
    for (int vv = 0; vv < 8; ++vv) {
      const int m = vv + half * 8;
      out[((size_t)(B0 + mt * 16 + m) * PRED + p) * CIN + c] = accs[n][vv] + bias;
    }
  }
}

// ---------------------------------------------------------------------------
extern "C" void kernel_launch(void* const* d_in, const int* in_sizes, int n_in,
                              void* d_out, int out_size, void* d_ws, size_t ws_size,
                              hipStream_t stream) {
  const float* x   = (const float*)d_in[0];
  const float* Ws  = (const float*)d_in[1];
  const float* bss = (const float*)d_in[2];
  const float* Wt  = (const float*)d_in[3];
  const float* bst = (const float*)d_in[4];
  float* out = (float*)d_out;

  // workspace layout (176 MB total)
  const size_t XT_BYTES = (size_t)CIN * NBATCH * SEQ * sizeof(float);  // 128 MB
  const size_t WELEMS   = (size_t)CIN * PRED * SEQ;                    // 6.29 M
  char* ws = (char*)d_ws;
  float*  xT     = (float*)ws;
  __bf16* WsT_hi = (__bf16*)(ws + XT_BYTES);
  __bf16* WsT_lo = WsT_hi + WELEMS;
  __bf16* WdT_hi = WsT_lo + WELEMS;
  __bf16* WdT_lo = WdT_hi + WELEMS;

  k1_prep_weights<<<dim3(SEQ / 64, CIN), 256, 0, stream>>>(
      Ws, Wt, WsT_hi, WsT_lo, WdT_hi, WdT_lo);
  k2_transpose_x<<<dim3(SEQ / 64, NBATCH), 256, 0, stream>>>(x, xT);
  k3_dlinear_gemm<<<dim3(NBATCH / BM, CIN), 256, 0, stream>>>(
      xT, WsT_hi, WsT_lo, WdT_hi, WdT_lo, bss, bst, out);
}
